// SSASparseTransformerCrossBlock_45689862095242
// MI455X (gfx1250) — matrix operand order repaired and checked by reading, not compiled, not run on hardware
//
#include <hip/hip_runtime.h>
#include <hip/hip_bf16.h>

typedef __bf16 bf16_t;
typedef __attribute__((ext_vector_type(16))) __bf16 v16bf;
typedef __attribute__((ext_vector_type(8)))  __bf16 v8bf;
typedef __attribute__((ext_vector_type(8)))  float  v8f;
typedef __attribute__((ext_vector_type(4)))  unsigned int v4u;
typedef __attribute__((ext_vector_type(8)))  int v8i;
typedef __attribute__((ext_vector_type(4)))  int v4i;

#define T_TOK 16384
#define C_DIM 1024
#define H_HEADS 16
#define D_HEAD 64
#define P_POOL 512
#define L_CTX 1024
#define HID_DIM 4096

__device__ __forceinline__ v8f v8f_zero() {
    v8f z;
#pragma unroll
    for (int i = 0; i < 8; ++i) z[i] = 0.0f;
    return z;
}

__device__ __forceinline__ v8f wmma_bf16(v16bf a, v16bf b, v8f c) {
    return __builtin_amdgcn_wmma_f32_16x16x32_bf16(
        false, a, false, b, (short)0, c, false, false);
}

// two contiguous 16B chunks -> one 16-half fragment (2x ds/global b128)
__device__ __forceinline__ v16bf ld_frag(const bf16_t* p0, const bf16_t* p1) {
    v8bf lo = *(const v8bf*)p0;
    v8bf hi = *(const v8bf*)p1;
    v16bf r;
#pragma unroll
    for (int i = 0; i < 8; ++i) { r[i] = lo[i]; r[8 + i] = hi[i]; }
    return r;
}

// ---------------------------------------------------------------------------
// Tensor Data Mover: 2-D tile of 2-byte elements, global -> LDS, with row
// padding so the tile lands in a [rows][tile_w + pad] LDS layout.
// pad_icode: pad every 2^(code+1) DWORDs; pad_acode: insert (code+1) DWORDs.
// amdgpu-toolchain builtin arity: (v4u, v8i, v4i, v4i, v8i, i32 cpol)
// ---------------------------------------------------------------------------
__device__ __forceinline__ void tdm_load_tile_2d(unsigned int lds_addr,
                                                 const bf16_t* gsrc,
                                                 unsigned int tensor_w,
                                                 unsigned int tensor_h,
                                                 unsigned int stride,
                                                 unsigned int tile_w,
                                                 unsigned int tile_h,
                                                 unsigned int pad_icode,
                                                 unsigned int pad_acode) {
    unsigned long long ga = (unsigned long long)(size_t)gsrc;
    v4u g0;
    g0[0] = 1u;                                        // count=1 valid user D#
    g0[1] = lds_addr;                                  // LDS byte address
    g0[2] = (unsigned int)(ga & 0xffffffffu);
    g0[3] = (unsigned int)((ga >> 32) & 0x01ffffffu) | (2u << 30);   // type=2
    v8i g1;
    g1[0] = (int)((1u << 16)                           // data_size = 2 bytes
                | (1u << 20)                           // pad_enable
                | (pad_icode << 22)                    // pad_interval
                | (pad_acode << 25));                  // pad_amount
    g1[1] = (int)((tensor_w & 0xffffu) << 16);         // tensor_dim0[15:0]
    g1[2] = (int)(((tensor_w >> 16) & 0xffffu)
                | ((tensor_h & 0xffffu) << 16));
    g1[3] = (int)(((tensor_h >> 16) & 0xffffu)
                | ((tile_w & 0xffffu) << 16));         // tile_dim0
    g1[4] = (int)(tile_h & 0xffffu);                   // tile_dim1
    g1[5] = (int)stride;                               // tensor_dim0_stride
    g1[6] = 0;
    g1[7] = 0;
    v4i z4 = {0, 0, 0, 0};
    v8i z8 = {0, 0, 0, 0, 0, 0, 0, 0};
    __builtin_amdgcn_tensor_load_to_lds(g0, g1, z4, z4, z8, 0);
}

// ---------------------------------------------------------------------------
__global__ void k_f2bf(const float* __restrict__ in, bf16_t* __restrict__ out, size_t n) {
    size_t i = (size_t)blockIdx.x * blockDim.x + threadIdx.x;
    size_t stride = (size_t)gridDim.x * blockDim.x;
    for (; i < n; i += stride) out[i] = (bf16_t)in[i];
}

__global__ void k_zero(float* __restrict__ p, size_t n) {
    size_t i = (size_t)blockIdx.x * blockDim.x + threadIdx.x;
    size_t stride = (size_t)gridDim.x * blockDim.x;
    for (; i < n; i += stride) p[i] = 0.0f;
}

// m = silu(mod) @ Wm + bm
__global__ void k_mod(const float* __restrict__ mod, const float* __restrict__ Wm,
                      const float* __restrict__ bm, float* __restrict__ m) {
    __shared__ float sm[C_DIM];
    int tid = threadIdx.x;
    for (int i = tid; i < C_DIM; i += 256) {
        float v = mod[i];
        sm[i] = v / (1.0f + __expf(-v));
    }
    __syncthreads();
    int j = blockIdx.x * 256 + tid;
    if (j >= 9 * C_DIM) return;
    float acc = bm[j];
    for (int i = 0; i < C_DIM; ++i)
        acc += sm[i] * Wm[(size_t)i * (9 * C_DIM) + j];
    m[j] = acc;
}

// LayerNorm + modulate (+ optional pooled segment-sum)
__global__ __launch_bounds__(256)
void k_ln_mod(const float* __restrict__ x, const float* __restrict__ sc,
              const float* __restrict__ sh, bf16_t* __restrict__ out_bf,
              float* __restrict__ pool_sum, const int* __restrict__ inv) {
    __shared__ float red[256];
    int row = blockIdx.x;
    int tid = threadIdx.x;
    const float* xr = x + (size_t)row * C_DIM;
    float v[4];
    float s = 0.0f;
#pragma unroll
    for (int i = 0; i < 4; ++i) { v[i] = xr[tid + 256 * i]; s += v[i]; }
    red[tid] = s; __syncthreads();
    for (int off = 128; off > 0; off >>= 1) {
        if (tid < off) red[tid] += red[tid + off];
        __syncthreads();
    }
    float mu = red[0] * (1.0f / C_DIM);
    __syncthreads();
    float s2 = 0.0f;
#pragma unroll
    for (int i = 0; i < 4; ++i) { float d = v[i] - mu; s2 += d * d; }
    red[tid] = s2; __syncthreads();
    for (int off = 128; off > 0; off >>= 1) {
        if (tid < off) red[tid] += red[tid + off];
        __syncthreads();
    }
    float rstd = rsqrtf(red[0] * (1.0f / C_DIM) + 1e-6f);
    int p = inv ? inv[row] : 0;
#pragma unroll
    for (int i = 0; i < 4; ++i) {
        int c = tid + 256 * i;
        float y = (v[i] - mu) * rstd * (1.0f + sc[c]) + sh[c];
        out_bf[(size_t)row * C_DIM + c] = (bf16_t)y;
        if (pool_sum) atomicAdd(&pool_sum[(size_t)p * C_DIM + c], y);
    }
}

__global__ void k_finishpool(const float* __restrict__ s, float* __restrict__ outF,
                             bf16_t* __restrict__ outB, size_t n) {
    size_t i = (size_t)blockIdx.x * blockDim.x + threadIdx.x;
    if (i >= n) return;
    float v = s[i] * (1.0f / 32.0f);
    if (outF) outF[i] = v;
    if (outB) outB[i] = (bf16_t)v;
}

__global__ void k_pool(const float* __restrict__ x, const int* __restrict__ inv,
                       float* __restrict__ pool_sum) {
    size_t idx = (size_t)blockIdx.x * blockDim.x + threadIdx.x;
    if (idx >= (size_t)T_TOK * C_DIM) return;
    int t = (int)(idx >> 10);
    int c = (int)(idx & 1023);
    atomicAdd(&pool_sum[(size_t)inv[t] * C_DIM + c], x[idx]);
}

// ---------------------------------------------------------------------------
// bf16 WMMA GEMM, 64x64 tile, 4 waves, double-buffered LDS.
// A tile: async TDM (overlapped with compute).  B tile: register-pipelined
// global loads, stored transposed so fragments are contiguous b128 reads.
// ---------------------------------------------------------------------------
__global__ __launch_bounds__(128)
void k_gemm(const bf16_t* __restrict__ A, const bf16_t* __restrict__ B,
            const float* __restrict__ bias, const float* __restrict__ gvec,
            const float* __restrict__ resid, float* __restrict__ outF,
            bf16_t* __restrict__ outB, int M, int N, int K) {
    __shared__ bf16_t As[2][64][40];   // TDM-padded rows
    __shared__ bf16_t Bt[2][64][40];   // transposed [col][K]

    int tid  = threadIdx.x;
    int wave = tid >> 5;
    int lane = tid & 31;
    int lhalf = lane >> 4;
    int ln = lane & 15;

    int m0 = blockIdx.y * 64;
    int n0 = blockIdx.x * 64;
    int wr = (wave >> 1) * 32;
    int wc = (wave & 1) * 32;
    int kb  = lhalf * 8;
    int kb2 = lhalf * 16;

    unsigned int as_lds[2] = { (unsigned int)(size_t)&As[0][0][0],
                               (unsigned int)(size_t)&As[1][0][0] };

    int br = tid >> 4;             // this thread's B stage row  (0..7 step via jj)
    int bc = (tid & 15) * 2;       // B stage col pair (reused below with jj)
    (void)br; (void)bc;

    v8f acc[2][2];
#pragma unroll
    for (int ti = 0; ti < 2; ++ti)
#pragma unroll
        for (int tj = 0; tj < 2; ++tj) acc[ti][tj] = v8f_zero();

    // ---- prologue: fill buffer 0 ----
    if (wave == 0) {
        tdm_load_tile_2d(as_lds[0], A + (size_t)m0 * K,
                         (unsigned int)K, (unsigned int)M, (unsigned int)K,
                         32u, 64u, 3u, 3u);
    }
    {
        unsigned int btmp[8];
#pragma unroll
        for (int jj = 0; jj < 8; ++jj) {
            int i = tid + jj * 128;
            int r = i >> 5, c = (i & 31) * 2;
            btmp[jj] = *(const unsigned int*)&B[(size_t)r * N + n0 + c];
        }
#pragma unroll
        for (int jj = 0; jj < 8; ++jj) {
            int i = tid + jj * 128;
            int r = i >> 5, c = (i & 31) * 2;
            const bf16_t* h2 = (const bf16_t*)&btmp[jj];
            Bt[0][c][r]     = h2[0];
            Bt[0][c + 1][r] = h2[1];
        }
    }
    if (wave == 0) __builtin_amdgcn_s_wait_tensorcnt(0);
    __syncthreads();

    int cur = 0;
    for (int k0 = 0; k0 < K; k0 += 32) {
        int nxt = cur ^ 1;
        bool has_next = (k0 + 32 < K);

        unsigned int btmp[8];
        if (has_next) {
            if (wave == 0) {
                tdm_load_tile_2d(as_lds[nxt], A + (size_t)m0 * K + k0 + 32,
                                 (unsigned int)K, (unsigned int)M, (unsigned int)K,
                                 32u, 64u, 3u, 3u);
            }
#pragma unroll
            for (int jj = 0; jj < 8; ++jj) {
                int i = tid + jj * 128;
                int r = i >> 5, c = (i & 31) * 2;
                btmp[jj] = *(const unsigned int*)&B[(size_t)(k0 + 32 + r) * N + n0 + c];
            }
            if (k0 + 64 < K)
                __builtin_prefetch(&B[(size_t)(k0 + 64 + (tid & 31)) * N + n0], 0, 1);
        }

        // compute on current buffer (copy engines run underneath)
        v16bf af[2], bfr[2];
#pragma unroll
        for (int ti = 0; ti < 2; ++ti) {
            const bf16_t* ap = &As[cur][wr + ti * 16 + ln][0];
            af[ti] = ld_frag(ap + kb, ap + 16 + kb);
        }
#pragma unroll
        for (int tj = 0; tj < 2; ++tj) {
            const bf16_t* bp = &Bt[cur][wc + tj * 16 + ln][0];
            bfr[tj] = ld_frag(bp + kb, bp + 16 + kb);
        }
#pragma unroll
        for (int ti = 0; ti < 2; ++ti)
#pragma unroll
            for (int tj = 0; tj < 2; ++tj)
                acc[ti][tj] = wmma_bf16(af[ti], bfr[tj], acc[ti][tj]);

        if (has_next) {
#pragma unroll
            for (int jj = 0; jj < 8; ++jj) {
                int i = tid + jj * 128;
                int r = i >> 5, c = (i & 31) * 2;
                const bf16_t* h2 = (const bf16_t*)&btmp[jj];
                Bt[nxt][c][r]     = h2[0];
                Bt[nxt][c + 1][r] = h2[1];
            }
            if (wave == 0) __builtin_amdgcn_s_wait_tensorcnt(0);
        }
        __syncthreads();
        cur = nxt;
    }

#pragma unroll
    for (int ti = 0; ti < 2; ++ti) {
#pragma unroll
        for (int tj = 0; tj < 2; ++tj) {
            int n = n0 + wc + tj * 16 + ln;
#pragma unroll
            for (int r = 0; r < 8; ++r) {
                int m = m0 + wr + ti * 16 + r + lhalf * 8;
                float v = acc[ti][tj][r];
                if (bias)  v += bias[n];
                if (gvec)  v *= gvec[n];
                if (resid) v += resid[(size_t)m * N + n];
                if (outF) outF[(size_t)m * N + n] = v;
                if (outB) outB[(size_t)m * N + n] = (bf16_t)v;
            }
        }
    }
}

// ---------------------------------------------------------------------------
// RoPE + l2norm + gamma*sqrt(D)
// ---------------------------------------------------------------------------
__global__ void k_ropeq(const float* __restrict__ qf, const int* __restrict__ coords,
                        const float* __restrict__ gamma, bf16_t* __restrict__ qb) {
    int th = blockIdx.x * 256 + threadIdx.x;
    if (th >= T_TOK * H_HEADS) return;
    int t = th >> 4, h = th & 15;
    float cs[32], sn[32];
#pragma unroll
    for (int j = 0; j < 32; ++j) { cs[j] = 1.0f; sn[j] = 0.0f; }
    for (int c = 0; c < 3; ++c) {
        float coord = (float)coords[t * 3 + c];
#pragma unroll
        for (int i = 0; i < 10; ++i) {
            float f = __powf(10000.0f, -(float)i * 0.1f);
            float a = coord * f;
            cs[c * 10 + i] = __cosf(a);
            sn[c * 10 + i] = __sinf(a);
        }
    }
    const float* qp = qf + (size_t)th * 64;
    float val[64];
#pragma unroll
    for (int j = 0; j < 32; ++j) {
        float xe = qp[2 * j], xo = qp[2 * j + 1];
        val[2 * j]     = xe * cs[j] - xo * sn[j];
        val[2 * j + 1] = xe * sn[j] + xo * cs[j];
    }
    float nrm = 0.0f;
#pragma unroll
    for (int d = 0; d < 64; ++d) nrm += val[d] * val[d];
    float invn = 8.0f / fmaxf(sqrtf(nrm), 1e-12f);
#pragma unroll
    for (int d = 0; d < 64; ++d)
        qb[(size_t)th * 64 + d] = (bf16_t)(val[d] * invn * gamma[h * 64 + d]);
}

__global__ void k_ropek(const float* __restrict__ kvf, const int* __restrict__ pcoords,
                        const float* __restrict__ gamma, bf16_t* __restrict__ kb,
                        bf16_t* __restrict__ vb) {
    int th = blockIdx.x * 256 + threadIdx.x;
    if (th >= P_POOL * H_HEADS) return;
    int p = th >> 4, h = th & 15;
    float cs[32], sn[32];
#pragma unroll
    for (int j = 0; j < 32; ++j) { cs[j] = 1.0f; sn[j] = 0.0f; }
    for (int c = 0; c < 3; ++c) {
        float coord = (float)(pcoords[p * 3 + c] * 4);
#pragma unroll
        for (int i = 0; i < 10; ++i) {
            float f = __powf(10000.0f, -(float)i * 0.1f);
            float a = coord * f;
            cs[c * 10 + i] = __cosf(a);
            sn[c * 10 + i] = __sinf(a);
        }
    }
    const float* kp = kvf + (size_t)p * 2048 + h * 64;
    const float* vp = kvf + (size_t)p * 2048 + 1024 + h * 64;
    float val[64];
#pragma unroll
    for (int j = 0; j < 32; ++j) {
        float xe = kp[2 * j], xo = kp[2 * j + 1];
        val[2 * j]     = xe * cs[j] - xo * sn[j];
        val[2 * j + 1] = xe * sn[j] + xo * cs[j];
    }
    float nrm = 0.0f;
#pragma unroll
    for (int d = 0; d < 64; ++d) nrm += val[d] * val[d];
    float invn = 8.0f / fmaxf(sqrtf(nrm), 1e-12f);
#pragma unroll
    for (int d = 0; d < 64; ++d) {
        kb[(size_t)th * 64 + d] = (bf16_t)(val[d] * invn * gamma[h * 64 + d]);
        vb[(size_t)th * 64 + d] = (bf16_t)vp[d];
    }
}

__global__ void k_splitckv(const float* __restrict__ ckv, bf16_t* __restrict__ ck,
                           bf16_t* __restrict__ cv) {
    size_t idx = (size_t)blockIdx.x * blockDim.x + threadIdx.x;
    if (idx >= (size_t)L_CTX * C_DIM) return;
    size_t l = idx >> 10, c = idx & 1023;
    ck[idx] = (bf16_t)ckv[l * 2048 + c];
    cv[idx] = (bf16_t)ckv[l * 2048 + 1024 + c];
}

// ---------------------------------------------------------------------------
// Flash attention via WMMA, double-buffered: K tiles via TDM ([key][64+8pad]),
// V tiles register-pipelined + stored transposed. One barrier per 32-key step.
// block = 256 threads (8 waves) = 128 queries; one head per blockIdx.y
// ---------------------------------------------------------------------------
__global__ __launch_bounds__(256)
void k_attn(const bf16_t* __restrict__ Q, const bf16_t* __restrict__ Kb,
            const bf16_t* __restrict__ Vb, bf16_t* __restrict__ O,
            int TK, float scale) {
    __shared__ bf16_t Ks[2][32][72];     // [key][d], TDM-padded (64 + 8)
    __shared__ bf16_t Vt[2][64][40];     // [d][key]
    __shared__ bf16_t PSs[8][16][40];    // per-wave P tile (bf16)

    int head = blockIdx.y;
    int tid  = threadIdx.x;
    int wave = tid >> 5;
    int lane = tid & 31;
    int lhalf = lane >> 4;
    int ln = lane & 15;
    int q0 = blockIdx.x * 128 + wave * 16;
    int kb  = lhalf * 8;
    int kb2 = lhalf * 16;

    unsigned int ks_lds[2] = { (unsigned int)(size_t)&Ks[0][0][0],
                               (unsigned int)(size_t)&Ks[1][0][0] };

    const bf16_t* qrow = Q + ((size_t)(q0 + ln) * H_HEADS + head) * 64;
    v16bf aq0 = ld_frag(qrow + kb,      qrow + 16 + kb);
    v16bf aq1 = ld_frag(qrow + 32 + kb, qrow + 48 + kb);

    float mrun[8], lsum[8];
    v8f o[4];
#pragma unroll
    for (int r = 0; r < 8; ++r) { mrun[r] = -1e30f; lsum[r] = 0.0f; }
#pragma unroll
    for (int dc = 0; dc < 4; ++dc) o[dc] = v8f_zero();

    // ---- prologue: stage key tile 0 ----
    if (wave == 0) {
        tdm_load_tile_2d(ks_lds[0], Kb + (size_t)head * 64,
                         64u, 32u, (unsigned int)(H_HEADS * 64),
                         64u, 32u, 4u, 3u);
    }
    {
        unsigned int vtmp[4];
#pragma unroll
        for (int jj = 0; jj < 4; ++jj) {
            int i = tid + jj * 256;
            int key = i >> 5, c = (i & 31) * 2;
            vtmp[jj] = *(const unsigned int*)
                (Vb + ((size_t)key * H_HEADS + head) * 64 + c);
        }
#pragma unroll
        for (int jj = 0; jj < 4; ++jj) {
            int i = tid + jj * 256;
            int key = i >> 5, c = (i & 31) * 2;
            const bf16_t* hv = (const bf16_t*)&vtmp[jj];
            Vt[0][c][key]     = hv[0];
            Vt[0][c + 1][key] = hv[1];
        }
    }
    if (wave == 0) __builtin_amdgcn_s_wait_tensorcnt(0);
    __syncthreads();

    int cur = 0;
    for (int kt = 0; kt < TK; kt += 32) {
        int nxt = cur ^ 1;
        bool has_next = (kt + 32 < TK);

        unsigned int vtmp[4];
        if (has_next) {
            if (wave == 0) {
                tdm_load_tile_2d(ks_lds[nxt],
                                 Kb + ((size_t)(kt + 32) * H_HEADS + head) * 64,
                                 64u, 32u, (unsigned int)(H_HEADS * 64),
                                 64u, 32u, 4u, 3u);
            }
#pragma unroll
            for (int jj = 0; jj < 4; ++jj) {
                int i = tid + jj * 256;
                int key = i >> 5, c = (i & 31) * 2;
                vtmp[jj] = *(const unsigned int*)
                    (Vb + ((size_t)(kt + 32 + key) * H_HEADS + head) * 64 + c);
            }
        }

        // ---- scores on current buffer ----
        v8f s[2];
#pragma unroll
        for (int c = 0; c < 2; ++c) {
            s[c] = v8f_zero();
            const bf16_t* krow = &Ks[cur][c * 16 + ln][0];
            v16bf b0 = ld_frag(krow + kb,      krow + 16 + kb);
            v16bf b1 = ld_frag(krow + 32 + kb, krow + 48 + kb);
            s[c] = wmma_bf16(aq0, b0, s[c]);
            s[c] = wmma_bf16(aq1, b1, s[c]);
        }
        // online softmax
#pragma unroll
        for (int r = 0; r < 8; ++r) {
            float v0 = s[0][r] * scale;
            float v1 = s[1][r] * scale;
            float t = fmaxf(v0, v1);
            t = fmaxf(t, __shfl_xor(t, 1, 32));
            t = fmaxf(t, __shfl_xor(t, 2, 32));
            t = fmaxf(t, __shfl_xor(t, 4, 32));
            t = fmaxf(t, __shfl_xor(t, 8, 32));
            float mn = fmaxf(mrun[r], t);
            float corr = __expf(mrun[r] - mn);
            float p0 = __expf(v0 - mn);
            float p1 = __expf(v1 - mn);
            float ps = p0 + p1;
            ps += __shfl_xor(ps, 1, 32);
            ps += __shfl_xor(ps, 2, 32);
            ps += __shfl_xor(ps, 4, 32);
            ps += __shfl_xor(ps, 8, 32);
            lsum[r] = lsum[r] * corr + ps;
            mrun[r] = mn;
#pragma unroll
            for (int dc = 0; dc < 4; ++dc) o[dc][r] *= corr;
            int m = r + lhalf * 8;
            PSs[wave][m][ln]      = (bf16_t)p0;
            PSs[wave][m][16 + ln] = (bf16_t)p1;
        }
        asm volatile("s_wait_dscnt 0x0" ::: "memory");  // same-wave LDS RAW fence

        const bf16_t* prow = &PSs[wave][ln][0];
        v16bf pa = ld_frag(prow + kb, prow + 16 + kb);
#pragma unroll
        for (int dc = 0; dc < 4; ++dc) {
            const bf16_t* vrow = &Vt[cur][dc * 16 + ln][0];
            v16bf bv = ld_frag(vrow + kb, vrow + 16 + kb);
            o[dc] = wmma_bf16(pa, bv, o[dc]);
        }

        // ---- drain pipelined V stores, publish next buffer ----
        if (has_next) {
#pragma unroll
            for (int jj = 0; jj < 4; ++jj) {
                int i = tid + jj * 256;
                int key = i >> 5, c = (i & 31) * 2;
                const bf16_t* hv = (const bf16_t*)&vtmp[jj];
                Vt[nxt][c][key]     = hv[0];
                Vt[nxt][c + 1][key] = hv[1];
            }
            if (wave == 0) __builtin_amdgcn_s_wait_tensorcnt(0);
        }
        __syncthreads();
        cur = nxt;
    }

#pragma unroll
    for (int dc = 0; dc < 4; ++dc) {
#pragma unroll
        for (int r = 0; r < 8; ++r) {
            float val = o[dc][r] / lsum[r];
            int m = q0 + r + lhalf * 8;
            O[((size_t)m * H_HEADS + head) * 64 + dc * 16 + ln] = (bf16_t)val;
        }
    }
}

__global__ void k_silu_gate(const float* __restrict__ ab, bf16_t* __restrict__ g) {
    size_t idx = (size_t)blockIdx.x * blockDim.x + threadIdx.x;
    if (idx >= (size_t)P_POOL * HID_DIM) return;
    size_t p = idx >> 12, j = idx & 4095;
    float a = ab[p * 8192 + j];
    float b = ab[p * 8192 + 4096 + j];
    g[idx] = (bf16_t)((a / (1.0f + __expf(-a))) * b);
}

__global__ void k_final(const float* __restrict__ x, const float* __restrict__ xp3,
                        const float* __restrict__ xp, const int* __restrict__ inv,
                        float* __restrict__ out) {
    size_t idx = (size_t)blockIdx.x * blockDim.x + threadIdx.x;
    if (idx >= (size_t)T_TOK * C_DIM) return;
    int t = (int)(idx >> 10);
    int c = (int)(idx & 1023);
    size_t pc = (size_t)inv[t] * C_DIM + c;
    out[idx] = x[idx] + xp3[pc] - xp[pc];
}

// ---------------------------------------------------------------------------
extern "C" void kernel_launch(void* const* d_in, const int* in_sizes, int n_in,
                              void* d_out, int out_size, void* d_ws, size_t ws_size,
                              hipStream_t stream) {
    (void)in_sizes; (void)n_in; (void)out_size; (void)ws_size;

    const float* feats  = (const float*)d_in[0];
    const float* modv   = (const float*)d_in[1];
    const float* ctx    = (const float*)d_in[2];
    const float* Wq     = (const float*)d_in[3];
    const float* bq     = (const float*)d_in[4];
    const float* Wkv    = (const float*)d_in[5];
    const float* bkv    = (const float*)d_in[6];
    const float* Wo     = (const float*)d_in[7];
    const float* bo     = (const float*)d_in[8];
    const float* qg     = (const float*)d_in[9];
    const float* kg     = (const float*)d_in[10];
    const float* Wcq    = (const float*)d_in[11];
    const float* bcq    = (const float*)d_in[12];
    const float* Wckv   = (const float*)d_in[13];
    const float* bckv   = (const float*)d_in[14];
    const float* Wco    = (const float*)d_in[15];
    const float* bco    = (const float*)d_in[16];
    const float* W12    = (const float*)d_in[17];
    const float* b12    = (const float*)d_in[18];
    const float* W3     = (const float*)d_in[19];
    const float* b3     = (const float*)d_in[20];
    const float* Wm     = (const float*)d_in[21];
    const float* bm     = (const float*)d_in[22];
    const int*   coords = (const int*)d_in[23];
    const int*   inv    = (const int*)d_in[24];
    const int*   pcoords= (const int*)d_in[25];
    float* out = (float*)d_out;

    char* ws = (char*)d_ws;
    size_t off = 0;
    auto alloc = [&](size_t bytes) -> char* {
        char* p = ws + off;
        off += (bytes + 255) & ~(size_t)255;
        return p;
    };
    float*  m_mod   = (float*)alloc(9 * C_DIM * 4);
    bf16_t* Wq_bf   = (bf16_t*)alloc((size_t)C_DIM * C_DIM * 2);
    bf16_t* Wkv_bf  = (bf16_t*)alloc((size_t)C_DIM * 2 * C_DIM * 2);
    bf16_t* Wo_bf   = (bf16_t*)alloc((size_t)C_DIM * C_DIM * 2);
    bf16_t* Wcq_bf  = (bf16_t*)alloc((size_t)C_DIM * C_DIM * 2);
    bf16_t* Wckv_bf = (bf16_t*)alloc((size_t)C_DIM * 2 * C_DIM * 2);
    bf16_t* Wco_bf  = (bf16_t*)alloc((size_t)C_DIM * C_DIM * 2);
    bf16_t* W12_bf  = (bf16_t*)alloc((size_t)C_DIM * 2 * HID_DIM * 2);
    bf16_t* W3_bf   = (bf16_t*)alloc((size_t)HID_DIM * C_DIM * 2);
    bf16_t* ctx_bf  = (bf16_t*)alloc((size_t)L_CTX * C_DIM * 2);
    bf16_t* h_bf    = (bf16_t*)alloc((size_t)T_TOK * C_DIM * 2);   // reused as o_bf
    float*  hp_sum  = (float*)alloc((size_t)P_POOL * C_DIM * 4);
    bf16_t* hp_bf   = (bf16_t*)alloc((size_t)P_POOL * C_DIM * 2);
    float*  qf      = (float*)alloc((size_t)T_TOK * C_DIM * 4);    // reused as x
    bf16_t* q_bf    = (bf16_t*)alloc((size_t)T_TOK * C_DIM * 2);
    float*  kvf     = (float*)alloc((size_t)P_POOL * 2 * C_DIM * 4);
    bf16_t* k_bf    = (bf16_t*)alloc((size_t)P_POOL * C_DIM * 2);
    bf16_t* v_bf    = (bf16_t*)alloc((size_t)P_POOL * C_DIM * 2);
    float*  xp_sum  = (float*)alloc((size_t)P_POOL * C_DIM * 4);
    float*  xp      = (float*)alloc((size_t)P_POOL * C_DIM * 4);
    bf16_t* h2_bf   = (bf16_t*)alloc((size_t)P_POOL * C_DIM * 2);
    bf16_t* cq_bf   = (bf16_t*)alloc((size_t)P_POOL * C_DIM * 2);
    float*  ckvf    = (float*)alloc((size_t)L_CTX * 2 * C_DIM * 4);
    bf16_t* ck_bf   = (bf16_t*)alloc((size_t)L_CTX * C_DIM * 2);
    bf16_t* cv_bf   = (bf16_t*)alloc((size_t)L_CTX * C_DIM * 2);
    bf16_t* co_bf   = (bf16_t*)alloc((size_t)P_POOL * C_DIM * 2);
    float*  xp2     = (float*)alloc((size_t)P_POOL * C_DIM * 4);
    bf16_t* h3_bf   = (bf16_t*)alloc((size_t)P_POOL * C_DIM * 2);
    float*  ab      = (float*)alloc((size_t)P_POOL * 2 * HID_DIM * 4);
    bf16_t* gate_bf = (bf16_t*)alloc((size_t)P_POOL * HID_DIM * 2);
    float*  xp3     = (float*)alloc((size_t)P_POOL * C_DIM * 4);
    bf16_t* o_bf = h_bf;
    float*  x    = qf;

    auto cvt = [&](const float* src, bf16_t* dst, size_t n) {
        int blocks = (int)((n + 255) / 256);
        if (blocks > 32768) blocks = 32768;
        hipLaunchKernelGGL(k_f2bf, dim3(blocks), dim3(256), 0, stream, src, dst, n);
    };

    cvt(Wq,   Wq_bf,   (size_t)C_DIM * C_DIM);
    cvt(Wkv,  Wkv_bf,  (size_t)C_DIM * 2 * C_DIM);
    cvt(Wo,   Wo_bf,   (size_t)C_DIM * C_DIM);
    cvt(Wcq,  Wcq_bf,  (size_t)C_DIM * C_DIM);
    cvt(Wckv, Wckv_bf, (size_t)C_DIM * 2 * C_DIM);
    cvt(Wco,  Wco_bf,  (size_t)C_DIM * C_DIM);
    cvt(W12,  W12_bf,  (size_t)C_DIM * 2 * HID_DIM);
    cvt(W3,   W3_bf,   (size_t)HID_DIM * C_DIM);
    cvt(ctx,  ctx_bf,  (size_t)L_CTX * C_DIM);

    hipLaunchKernelGGL(k_mod, dim3(36), dim3(256), 0, stream, modv, Wm, bm, m_mod);
    hipLaunchKernelGGL(k_zero, dim3(2048), dim3(256), 0, stream, hp_sum, (size_t)P_POOL * C_DIM);
    hipLaunchKernelGGL(k_zero, dim3(2048), dim3(256), 0, stream, xp_sum, (size_t)P_POOL * C_DIM);

    hipLaunchKernelGGL(k_ln_mod, dim3(T_TOK), dim3(256), 0, stream,
                       feats, m_mod + C_DIM, m_mod + 0, h_bf, hp_sum, inv);
    hipLaunchKernelGGL(k_finishpool, dim3((P_POOL * C_DIM + 255) / 256), dim3(256), 0, stream,
                       hp_sum, (float*)nullptr, hp_bf, (size_t)P_POOL * C_DIM);

    hipLaunchKernelGGL(k_gemm, dim3(C_DIM / 64, T_TOK / 64), dim3(128), 0, stream,
                       h_bf, Wq_bf, bq, (const float*)nullptr, (const float*)nullptr,
                       qf, (bf16_t*)nullptr, T_TOK, C_DIM, C_DIM);
    hipLaunchKernelGGL(k_gemm, dim3(2 * C_DIM / 64, P_POOL / 64), dim3(128), 0, stream,
                       hp_bf, Wkv_bf, bkv, (const float*)nullptr, (const float*)nullptr,
                       kvf, (bf16_t*)nullptr, P_POOL, 2 * C_DIM, C_DIM);

    hipLaunchKernelGGL(k_ropeq, dim3(T_TOK * H_HEADS / 256), dim3(256), 0, stream,
                       qf, coords, qg, q_bf);
    hipLaunchKernelGGL(k_ropek, dim3(P_POOL * H_HEADS / 256), dim3(256), 0, stream,
                       kvf, pcoords, kg, k_bf, v_bf);

    hipLaunchKernelGGL(k_attn, dim3(T_TOK / 128, H_HEADS), dim3(256), 0, stream,
                       q_bf, k_bf, v_bf, o_bf, P_POOL, 0.125f);

    hipLaunchKernelGGL(k_gemm, dim3(C_DIM / 64, T_TOK / 64), dim3(128), 0, stream,
                       o_bf, Wo_bf, bo, m_mod + 2 * C_DIM, feats,
                       x, (bf16_t*)nullptr, T_TOK, C_DIM, C_DIM);

    hipLaunchKernelGGL(k_pool, dim3((int)(((size_t)T_TOK * C_DIM + 255) / 256)), dim3(256), 0,
                       stream, x, inv, xp_sum);
    hipLaunchKernelGGL(k_finishpool, dim3((P_POOL * C_DIM + 255) / 256), dim3(256), 0, stream,
                       xp_sum, xp, (bf16_t*)nullptr, (size_t)P_POOL * C_DIM);

    hipLaunchKernelGGL(k_ln_mod, dim3(P_POOL), dim3(256), 0, stream,
                       xp, m_mod + 4 * C_DIM, m_mod + 3 * C_DIM, h2_bf,
                       (float*)nullptr, (const int*)nullptr);

    hipLaunchKernelGGL(k_gemm, dim3(C_DIM / 64, P_POOL / 64), dim3(128), 0, stream,
                       h2_bf, Wcq_bf, bcq, (const float*)nullptr, (const float*)nullptr,
                       (float*)nullptr, cq_bf, P_POOL, C_DIM, C_DIM);
    hipLaunchKernelGGL(k_gemm, dim3(2 * C_DIM / 64, L_CTX / 64), dim3(128), 0, stream,
                       ctx_bf, Wckv_bf, bckv, (const float*)nullptr, (const float*)nullptr,
                       ckvf, (bf16_t*)nullptr, L_CTX, 2 * C_DIM, C_DIM);
    hipLaunchKernelGGL(k_splitckv, dim3((int)(((size_t)L_CTX * C_DIM + 255) / 256)), dim3(256),
                       0, stream, ckvf, ck_bf, cv_bf);

    hipLaunchKernelGGL(k_attn, dim3(P_POOL / 128, H_HEADS), dim3(256), 0, stream,
                       cq_bf, ck_bf, cv_bf, co_bf, L_CTX, 0.125f);

    hipLaunchKernelGGL(k_gemm, dim3(C_DIM / 64, P_POOL / 64), dim3(128), 0, stream,
                       co_bf, Wco_bf, bco, m_mod + 5 * C_DIM, xp,
                       xp2, (bf16_t*)nullptr, P_POOL, C_DIM, C_DIM);

    hipLaunchKernelGGL(k_ln_mod, dim3(P_POOL), dim3(256), 0, stream,
                       xp2, m_mod + 7 * C_DIM, m_mod + 6 * C_DIM, h3_bf,
                       (float*)nullptr, (const int*)nullptr);
    hipLaunchKernelGGL(k_gemm, dim3(2 * HID_DIM / 64, P_POOL / 64), dim3(128), 0, stream,
                       h3_bf, W12_bf, b12, (const float*)nullptr, (const float*)nullptr,
                       ab, (bf16_t*)nullptr, P_POOL, 2 * HID_DIM, C_DIM);
    hipLaunchKernelGGL(k_silu_gate, dim3((int)(((size_t)P_POOL * HID_DIM + 255) / 256)),
                       dim3(256), 0, stream, ab, gate_bf);
    hipLaunchKernelGGL(k_gemm, dim3(C_DIM / 64, P_POOL / 64), dim3(128), 0, stream,
                       gate_bf, W3_bf, b3, m_mod + 8 * C_DIM, xp2,
                       xp3, (bf16_t*)nullptr, P_POOL, C_DIM, HID_DIM);

    hipLaunchKernelGGL(k_final, dim3((int)(((size_t)T_TOK * C_DIM + 255) / 256)), dim3(256),
                       0, stream, x, xp3, xp, inv, out);
}
